// TensorRT_Vanilla_317827580180
// MI455X (gfx1250) — matrix operand order repaired and checked by reading, not compiled, not run on hardware
//
#include <hip/hip_runtime.h>

// ---------------------------------------------------------------------------
// Problem constants (from reference): M=1024, N=4096, D=128, H=32, P=4096
// ---------------------------------------------------------------------------
#define MM   1024
#define NN   4096
#define DD   128
#define HH   32
#define PP   4096
#define PC   5120   // P + M

typedef __attribute__((ext_vector_type(16))) __bf16 v16bf;
typedef __attribute__((ext_vector_type(8)))  float  v8f;
typedef __attribute__((ext_vector_type(4)))  unsigned int v4u;
typedef __attribute__((ext_vector_type(8)))  int v8i;
typedef __attribute__((ext_vector_type(4)))  int v4i;

union ABfrag {
    v16bf        v;
    unsigned int d[8];
};

__device__ inline v8f zero8() {
    v8f z = {0.f, 0.f, 0.f, 0.f, 0.f, 0.f, 0.f, 0.f};
    return z;
}

// Native bf16 conversions -> v_cvt_pk_bf16_f32 class instructions
__device__ inline unsigned short f32_to_bf16(float f) {
    union { __bf16 h; unsigned short u; } c;
    c.h = (__bf16)f;
    return c.u;
}
__device__ inline unsigned int pack_bf16x2(float a, float b) {
    union { __bf16 h[2]; unsigned int u; } c;
    c.h[0] = (__bf16)a;
    c.h[1] = (__bf16)b;
    return c.u;
}

// wave32 xor-shuffle via ds_swizzle (group-of-32 mode: and=0x1F, or=0, xor=MASK)
template <int MASK>
__device__ inline float shfl_xor_f(float x) {
    int v = __builtin_amdgcn_ds_swizzle(__float_as_int(x), (MASK << 10) | 0x1F);
    return __int_as_float(v);
}

// ---------------------------------------------------------------------------
// CDNA5 async copy: global -> LDS, 16B per lane, tracked by ASYNCcnt
// ---------------------------------------------------------------------------
__device__ inline unsigned lds_addr32(const void* p) {
    // generic pointer into LDS aperture: addr[31:0] == LDS byte offset
    return (unsigned)(unsigned long long)p;
}
__device__ inline void async_ld_b128(unsigned ldsoff, const void* gptr) {
    asm volatile("global_load_async_to_lds_b128 %0, %1, off"
                 :: "v"(ldsoff), "v"(gptr)
                 : "memory");
}
__device__ inline void wait_asynccnt0() {
    asm volatile("s_wait_asynccnt 0x0" ::: "memory");
}

// ---------------------------------------------------------------------------
// CDNA5 Tensor Data Mover: 2-D bf16 tile, global -> LDS, tracked by TENSORcnt.
// Descriptor per ISA 8.3/8.4: group0 {count=1, lds_addr, global_addr, type=2},
// group1 {data_size=2B, tensor dims, tile dims, dim0 stride}; groups 2/3 zero
// (tile_dim2 = 0 -> 2-D). This toolchain exposes the 6-arg builtin form:
// (v4u g0, v8i g1, v4i g2, v4i g3, v8i extra, i32 cpol).
// ---------------------------------------------------------------------------
__device__ inline void tdm_load2d_bf16(unsigned ldsaddr, const void* g,
                                       unsigned td0, unsigned td1,
                                       unsigned tile0, unsigned tile1,
                                       unsigned stride0) {
    unsigned long long ga = (unsigned long long)g;
    v4u g0;
    g0[0] = 1u;                                                 // count=1, user
    g0[1] = ldsaddr;                                            // lds_addr
    g0[2] = (unsigned)ga;                                       // global_addr lo
    g0[3] = ((unsigned)(ga >> 32) & 0x01FFFFFFu) | (2u << 30);  // addr hi | type=2
    v8i g1;
    g1[0] = (int)(1u << 16);                                    // data_size=1 (2B)
    g1[1] = (int)((td0 & 0xFFFFu) << 16);                       // tensor_dim0 lo16
    g1[2] = (int)(((td0 >> 16) & 0xFFFFu) | ((td1 & 0xFFFFu) << 16));
    g1[3] = (int)(((td1 >> 16) & 0xFFFFu) | ((tile0 & 0xFFFFu) << 16));
    g1[4] = (int)(tile1 & 0xFFFFu);                             // tile_dim1, dim2=0
    g1[5] = (int)stride0;                                       // dim0 stride lo32
    g1[6] = 0;                                                  // stride hi / dim1 stride
    g1[7] = 0;
    v4i z4 = {0, 0, 0, 0};
    v8i z8 = {0, 0, 0, 0, 0, 0, 0, 0};
    __builtin_amdgcn_tensor_load_to_lds(g0, g1, z4, z4, z8, 0);
}
__device__ inline void wait_tensorcnt0() {
    __builtin_amdgcn_s_wait_tensorcnt((short)0);
}

// ---------------------------------------------------------------------------
// Conversion kernels
// ---------------------------------------------------------------------------
__global__ void cvt_x_kernel(const float* __restrict__ X,
                             unsigned short* __restrict__ Xb, int n) {
    int i = blockIdx.x * 256 + threadIdx.x;
    if (i < n) Xb[i] = f32_to_bf16(X[i]);
}

// cache_K [H][P][D] fp32  ->  Kt [H][D][PC] bf16 (transposed, p in [0,P))
__global__ void cvt_cacheK_kernel(const float* __restrict__ cK,
                                  unsigned short* __restrict__ Kt) {
    int i = blockIdx.x * 256 + threadIdx.x;      // over H*P*D = 16M
    int d = i & (DD - 1);
    int p = (i >> 7) & (PP - 1);
    int h = i >> 19;
    Kt[(h * DD + d) * PC + p] = f32_to_bf16(cK[i]);
}

// cache_V [H][P][D] fp32  ->  Vc [H][PC][D] bf16 (straight, p in [0,P))
__global__ void cvt_cacheV_kernel(const float* __restrict__ cV,
                                  unsigned short* __restrict__ Vc) {
    int i = blockIdx.x * 256 + threadIdx.x;
    int d = i & (DD - 1);
    int p = (i >> 7) & (PP - 1);
    int h = i >> 19;
    Vc[(h * PC + p) * DD + d] = f32_to_bf16(cV[i]);
}

// ---------------------------------------------------------------------------
// bf16 WMMA GEMM:  C[M,N] = Xb[M,K=4096] * W[K,N]   (W converted on the fly)
// Block: 256 threads = 8 waves, tile 64(M) x 256(N), K-step 32.
// Double-buffered LDS; A tile via async-to-LDS (ASYNCcnt), B tile fetched to
// VGPRs before compute and converted/stored after.
// MODE 0: -> Qb[h][m][d]    MODE 1: -> Kt[h][d][P+m]    MODE 2: -> Vc[h][P+m][d]
// ---------------------------------------------------------------------------
template <int MODE>
__global__ __launch_bounds__(256)
void gemm_xw_kernel(const unsigned short* __restrict__ Xb,
                    const float* __restrict__ W,
                    unsigned short* __restrict__ Out) {
    __shared__ __align__(16) unsigned short As[2][64 * 32];    // [m][k]
    __shared__ __align__(16) unsigned short Bs[2][32 * 256];   // [k][n]

    const int tid   = threadIdx.x;
    const int lane  = tid & 31;
    const int wid   = tid >> 5;
    const int waveM = wid >> 2;                 // 0..1
    const int waveN = wid & 3;                  // 0..3
    const int bm    = blockIdx.x * 64;
    const int bn    = blockIdx.y * 256;
    const int half  = lane >> 4;                // 0/1
    const int l16   = lane & 15;

    // A-tile mapping: one async b128 per thread per K-step
    const int ar = tid >> 2;                    // 0..63
    const int ak = (tid & 3) * 8;               // 0,8,16,24
    const unsigned short* aSrc = Xb + (bm + ar) * NN + ak;
    const unsigned aDst0 = lds_addr32(&As[0][ar * 32 + ak]);
    const unsigned aDst1 = lds_addr32(&As[1][ar * 32 + ak]);

    float4 wreg[8];
    auto fetchB = [&](int kb) {
        #pragma unroll
        for (int i = 0; i < 8; ++i) {
            int flat = tid + i * 256;           // float4 index in [0,2048)
            int row  = flat >> 6;
            int c4   = (flat & 63) * 4;
            wreg[i] = *(const float4*)(W + (kb + row) * NN + bn + c4);
        }
    };
    auto storeB = [&](int buf) {
        #pragma unroll
        for (int i = 0; i < 8; ++i) {
            int flat = tid + i * 256;
            int row  = flat >> 6;
            int c4   = (flat & 63) * 4;
            *(uint2*)(&Bs[buf][row * 256 + c4]) =
                make_uint2(pack_bf16x2(wreg[i].x, wreg[i].y),
                           pack_bf16x2(wreg[i].z, wreg[i].w));
        }
    };

    v8f acc[2][4];
    #pragma unroll
    for (int s = 0; s < 2; ++s)
        #pragma unroll
        for (int t = 0; t < 4; ++t) acc[s][t] = zero8();

    // ---- prologue: tile 0 into buffer 0
    fetchB(0);
    async_ld_b128(aDst0, aSrc);
    storeB(0);
    wait_asynccnt0();
    __syncthreads();

    int cur = 0;
    for (int kb = 0; kb < NN; kb += 32) {
        const bool hasNext = (kb + 32) < NN;
        if (hasNext) {
            // issue next tile: A via async-to-LDS, B into VGPRs
            async_ld_b128(cur ? aDst0 : aDst1, aSrc + kb + 32);
            fetchB(kb + 32);
            if (kb + 64 < NN)   // warm GL2 two steps ahead (speculative)
                __builtin_prefetch(
                    W + (kb + 64 + (tid >> 3)) * NN + bn + (tid & 7) * 32, 0, 0);
        }

        // ---- compute on buffer `cur`
        ABfrag afr[2];
        #pragma unroll
        for (int s = 0; s < 2; ++s) {
            int m = waveM * 32 + s * 16 + l16;
            const unsigned int* arow = (const unsigned int*)(&As[cur][m * 32]);
            #pragma unroll
            for (int j = 0; j < 8; ++j) {
                int k = (j < 4) ? (half * 8 + j * 2)
                                : (16 + half * 8 + (j - 4) * 2);
                afr[s].d[j] = arow[k >> 1];
            }
        }
        ABfrag bfr[4];
        #pragma unroll
        for (int t = 0; t < 4; ++t) {
            const unsigned short* bcol = &Bs[cur][waveN * 64 + t * 16 + l16];
            #pragma unroll
            for (int j = 0; j < 8; ++j) {
                int k = (j < 4) ? (half * 8 + j * 2)
                                : (16 + half * 8 + (j - 4) * 2);
                bfr[t].d[j] = (unsigned)bcol[k * 256] | ((unsigned)bcol[(k + 1) * 256] << 16);
            }
        }
        #pragma unroll
        for (int s = 0; s < 2; ++s)
            #pragma unroll
            for (int t = 0; t < 4; ++t)
                acc[s][t] = __builtin_amdgcn_wmma_f32_16x16x32_bf16(
                    false, afr[s].v, false, bfr[t].v, (short)0, acc[s][t],
                    false, false);

        // ---- convert + store next B tile
        if (hasNext) storeB(cur ^ 1);
        wait_asynccnt0();
        __syncthreads();
        cur ^= 1;
    }

    // ---- store: C layout — VGPR r: m = r + 8*half, n = l16
    #pragma unroll
    for (int s = 0; s < 2; ++s)
        #pragma unroll
        for (int t = 0; t < 4; ++t)
            #pragma unroll
            for (int r = 0; r < 8; ++r) {
                int m = bm + waveM * 32 + s * 16 + r + half * 8;
                int n = bn + waveN * 64 + t * 16 + l16;
                unsigned short val = f32_to_bf16(acc[s][t][r]);
                int h = n >> 7, d = n & (DD - 1);
                if (MODE == 0)
                    Out[(h * MM + m) * DD + d] = val;            // Qb
                else if (MODE == 1)
                    Out[(h * DD + d) * PC + PP + m] = val;       // Kt
                else
                    Out[(h * PC + (PP + m)) * DD + d] = val;     // Vc
            }
}

// ---------------------------------------------------------------------------
// Flash attention, block-cooperative: 8 waves of a block share one head and
// the same 32-key blocks. K/V tiles are staged into LDS by the Tensor Data
// Mover (one issuing wave, TENSORcnt + barrier), double-buffered — 8x less L2
// traffic than per-wave loads. Online softmax, no 1/sqrt(D) scale (matches
// reference).
// ---------------------------------------------------------------------------
__global__ __launch_bounds__(256)
void attn_kernel(const unsigned short* __restrict__ Qb,
                 const unsigned short* __restrict__ Kt,
                 const unsigned short* __restrict__ Vc,
                 float* __restrict__ Out) {
    __shared__ __align__(16) unsigned short Kls[2][DD * 32];   // [d][p_local]
    __shared__ __align__(16) unsigned short Vls[2][32 * DD];   // [p_local][d]
    __shared__ __align__(16) unsigned short Pst[8][16 * 32];   // per-wave P relayout

    const int tid  = threadIdx.x;
    const int lane = tid & 31;
    const int wid  = tid >> 5;
    const int h    = blockIdx.x >> 3;                 // head, uniform per block
    const int m0   = ((blockIdx.x & 7) * 8 + wid) * 16;
    const int half = lane >> 4;
    const int l16  = lane & 15;

    // Q fragments (16 rows x 128 d = 4 K-steps), kept resident all loop
    ABfrag qf[4];
    {
        const unsigned short* qrow = Qb + (h * MM + (m0 + l16)) * DD;
        #pragma unroll
        for (int f = 0; f < 4; ++f)
            #pragma unroll
            for (int j = 0; j < 8; ++j) {
                int k = f * 32 + ((j < 4) ? (half * 8 + j * 2)
                                          : (16 + half * 8 + (j - 4) * 2));
                qf[f].d[j] = *(const unsigned int*)(qrow + k);
            }
    }

    float mi[8], li[8];
    v8f o[8];
    #pragma unroll
    for (int r = 0; r < 8; ++r) { mi[r] = -1e30f; li[r] = 0.f; }
    #pragma unroll
    for (int f = 0; f < 8; ++f) o[f] = zero8();

    const unsigned short* Kth = Kt + h * DD * PC;
    const unsigned short* Vch = Vc + h * PC * DD;
    const unsigned kDst0 = lds_addr32(&Kls[0][0]);
    const unsigned kDst1 = lds_addr32(&Kls[1][0]);
    const unsigned vDst0 = lds_addr32(&Vls[0][0]);
    const unsigned vDst1 = lds_addr32(&Vls[1][0]);
    unsigned short* pst = Pst[wid];

    // ---- prologue: TDM fetch of key block 0 into buffer 0
    if (wid == 0) {
        // K tile: 32 contiguous keys x 128 d-rows from Kt[h][d][p], stride PC
        tdm_load2d_bf16(kDst0, Kth, PC, DD, 32, DD, PC);
        // V tile: 128 contiguous d x 32 key-rows from Vc[h][p][d], stride DD
        tdm_load2d_bf16(vDst0, Vch, DD, PC, DD, 32, DD);
        wait_tensorcnt0();
    }
    __syncthreads();

    int cur = 0;
    for (int pb = 0; pb < PC; pb += 32) {
        // ---- issue TDM for the next key block into the other buffer
        if ((pb + 32) < PC && wid == 0) {
            tdm_load2d_bf16(cur ? kDst0 : kDst1, Kth + (pb + 32),
                            PC, DD, 32, DD, PC);
            tdm_load2d_bf16(cur ? vDst0 : vDst1, Vch + (pb + 32) * DD,
                            DD, PC, DD, 32, DD);
        }

        // ---- scores: two 16x16 tiles, contraction over D=128 (4 WMMAs each)
        v8f S[2];
        #pragma unroll
        for (int t = 0; t < 2; ++t) {
            v8f sacc = zero8();
            #pragma unroll
            for (int f = 0; f < 4; ++f) {
                ABfrag kb;   // lane%16 = key, 16 d in table order (from LDS)
                const unsigned short* kcol = &Kls[cur][(f * 32) * 32 + t * 16 + l16];
                #pragma unroll
                for (int j = 0; j < 8; ++j) {
                    int k = (j < 4) ? (half * 8 + j * 2)
                                    : (16 + half * 8 + (j - 4) * 2);
                    kb.d[j] = (unsigned)kcol[k * 32] | ((unsigned)kcol[(k + 1) * 32] << 16);
                }
                sacc = __builtin_amdgcn_wmma_f32_16x16x32_bf16(
                    false, qf[f].v, false, kb.v, (short)0, sacc, false, false);
            }
            S[t] = sacc;
        }

        // ---- online softmax (row stats per (r, lane-half) via xor shuffles)
        float alpha[8];
        #pragma unroll
        for (int r = 0; r < 8; ++r) {
            float mx = fmaxf(S[0][r], S[1][r]);
            mx = fmaxf(mx, shfl_xor_f<1>(mx));
            mx = fmaxf(mx, shfl_xor_f<2>(mx));
            mx = fmaxf(mx, shfl_xor_f<4>(mx));
            mx = fmaxf(mx, shfl_xor_f<8>(mx));
            float newm = fmaxf(mi[r], mx);
            alpha[r] = __expf(mi[r] - newm);
            mi[r] = newm;
            float p0v = __expf(S[0][r] - newm);
            float p1v = __expf(S[1][r] - newm);
            S[0][r] = p0v; S[1][r] = p1v;
            float sum = p0v + p1v;
            sum += shfl_xor_f<1>(sum);
            sum += shfl_xor_f<2>(sum);
            sum += shfl_xor_f<4>(sum);
            sum += shfl_xor_f<8>(sum);
            li[r] = li[r] * alpha[r] + sum;
        }

        // ---- relayout P (C-frag) -> LDS -> A-frag (same-wave, dscnt-ordered)
        #pragma unroll
        for (int r = 0; r < 8; ++r) {
            int m = r + half * 8;
            pst[m * 32 + l16]      = f32_to_bf16(S[0][r]);
            pst[m * 32 + 16 + l16] = f32_to_bf16(S[1][r]);
        }
        ABfrag pa;
        {
            const unsigned int* prow = (const unsigned int*)(pst + l16 * 32);
            #pragma unroll
            for (int j = 0; j < 8; ++j) {
                int k = (j < 4) ? (half * 8 + j * 2)
                                : (16 + half * 8 + (j - 4) * 2);
                pa.d[j] = prow[k >> 1];
            }
        }

        // ---- O = diag(alpha)*O + P @ V  (8 WMMAs cover d = 0..127)
        #pragma unroll
        for (int f = 0; f < 8; ++f) {
            v8f of = o[f];
            #pragma unroll
            for (int r = 0; r < 8; ++r) of[r] *= alpha[r];
            ABfrag vb;   // lane%16 = d, 16 keys in table order (from LDS)
            const unsigned short* vcol = &Vls[cur][f * 16 + l16];
            #pragma unroll
            for (int j = 0; j < 8; ++j) {
                int k = (j < 4) ? (half * 8 + j * 2)
                                : (16 + half * 8 + (j - 4) * 2);
                vb.d[j] = (unsigned)vcol[k * DD] | ((unsigned)vcol[(k + 1) * DD] << 16);
            }
            o[f] = __builtin_amdgcn_wmma_f32_16x16x32_bf16(
                false, pa.v, false, vb.v, (short)0, of, false, false);
        }

        // ---- wait for the next-buffer DMA, then swap
        if (wid == 0) wait_tensorcnt0();
        __syncthreads();
        cur ^= 1;
    }

    // ---- epilogue: Out[m][h*128 + d] = o / l
    #pragma unroll
    for (int f = 0; f < 8; ++f)
        #pragma unroll
        for (int r = 0; r < 8; ++r) {
            int m   = m0 + r + half * 8;
            int col = h * DD + f * 16 + l16;
            Out[m * NN + col] = o[f][r] / li[r];
        }
}

// ---------------------------------------------------------------------------
// Host launcher
// ---------------------------------------------------------------------------
extern "C" void kernel_launch(void* const* d_in, const int* in_sizes, int n_in,
                              void* d_out, int out_size, void* d_ws, size_t ws_size,
                              hipStream_t stream) {
    const float* X  = (const float*)d_in[0];
    const float* Wq = (const float*)d_in[1];
    const float* Wk = (const float*)d_in[2];
    const float* Wv = (const float*)d_in[3];
    const float* cK = (const float*)d_in[4];
    const float* cV = (const float*)d_in[5];
    float* out = (float*)d_out;

    // Workspace layout (bf16 ushorts): Xb 8MB | Qb 8MB | Kt 40MB | Vc 40MB = 96MB
    unsigned short* ws = (unsigned short*)d_ws;
    unsigned short* Xb = ws;
    unsigned short* Qb = Xb + (size_t)MM * NN;            // 4,194,304
    unsigned short* Kt = Qb + (size_t)HH * MM * DD;       // +4,194,304
    unsigned short* Vc = Kt + (size_t)HH * DD * PC;       // +20,971,520

    cvt_x_kernel<<<(MM * NN) / 256, 256, 0, stream>>>(X, Xb, MM * NN);
    cvt_cacheK_kernel<<<(HH * PP * DD) / 256, 256, 0, stream>>>(cK, Kt);
    cvt_cacheV_kernel<<<(HH * PP * DD) / 256, 256, 0, stream>>>(cV, Vc);

    dim3 g(MM / 64, NN / 256);   // 16 x 16 blocks
    gemm_xw_kernel<0><<<g, 256, 0, stream>>>(Xb, Wq, Qb);
    gemm_xw_kernel<1><<<g, 256, 0, stream>>>(Xb, Wk, Kt);
    gemm_xw_kernel<2><<<g, 256, 0, stream>>>(Xb, Wv, Vc);

    attn_kernel<<<(HH * (MM / 16)) / 8, 256, 0, stream>>>(Qb, Kt, Vc, out);
}
